// HAttention1D_91010357002503
// MI455X (gfx1250) — matrix operand order repaired and checked by reading, not compiled, not run on hardware
//
#include <hip/hip_runtime.h>
#include <cstdint>
#include <cstddef>

// ---------- CDNA5 WMMA types ----------
typedef __attribute__((ext_vector_type(16))) __bf16 v16bf;
typedef __attribute__((ext_vector_type(8)))  float  v8f;
typedef unsigned int v4u __attribute__((ext_vector_type(4)));
typedef int v4i __attribute__((ext_vector_type(4)));
typedef int v8i __attribute__((ext_vector_type(8)));

union FragBF {
  v16bf bf;
  unsigned short u[16];
  uint4 q[2];
};

__device__ __forceinline__ unsigned short f2bf(float f) {
  unsigned int u = __float_as_uint(f);
  u += 0x7fffu + ((u >> 16) & 1u);   // round to nearest even
  return (unsigned short)(u >> 16);
}
__device__ __forceinline__ float bf2f(unsigned short h) {
  return __uint_as_float(((unsigned int)h) << 16);
}
__device__ __forceinline__ v8f wmma_bf16(const FragBF& a, const FragBF& b, v8f c) {
  return __builtin_amdgcn_wmma_f32_16x16x32_bf16(false, a.bf, false, b.bf,
                                                 (short)0, c, false, false);
}

// ---------- Tensor Data Mover: 2D bf16 tile -> LDS ----------
// D# per CDNA5 ISA.
// tile: tile_d1 rows x tile_d0 elements (2B each), row stride row_stride elems.
__device__ __forceinline__ void tdm_load_2d(unsigned lds_off, const void* gptr,
                                            unsigned tensor_d0, unsigned tensor_d1,
                                            unsigned tile_d0, unsigned tile_d1,
                                            unsigned row_stride) {
  unsigned long long ga = (unsigned long long)(uintptr_t)gptr;
  v4u g0;
  g0[0] = 1u;                                  // count=1, user mode
  g0[1] = lds_off;                             // lds_addr [63:32]
  g0[2] = (unsigned)ga;                        // global_addr [95:64]
  g0[3] = 0x80000000u |                        // type=2 at [127:126]
          ((unsigned)(ga >> 32) & 0x01FFFFFFu);
  v8i g1;
  g1[0] = (int)(1u << 16);                     // data_size=1 (2 bytes), wg_mask=0
  g1[1] = (int)((tensor_d0 & 0xFFFFu) << 16);  // tensor_dim0 [79:48] low half
  g1[2] = (int)(((tensor_d0 >> 16) & 0xFFFFu) |
                ((tensor_d1 & 0xFFFFu) << 16));// dim0 hi | tensor_dim1 lo
  g1[3] = (int)(((tensor_d1 >> 16) & 0xFFFFu) |
                (tile_d0 << 16));              // dim1 hi | tile_dim0 [127:112]
  g1[4] = (int)(tile_d1 & 0xFFFFu);            // tile_dim1 [143:128]; tile_dim2=0
  g1[5] = (int)row_stride;                     // tensor_dim0_stride [207:160] lo32
  g1[6] = 0;                                   // stride hi16 | dim1_stride lo
  g1[7] = 0;
  v4i z4 = {0, 0, 0, 0};
#if defined(__clang_major__) && (__clang_major__ >= 23)
  v8i z8 = {0, 0, 0, 0, 0, 0, 0, 0};
  __builtin_amdgcn_tensor_load_to_lds(g0, g1, z4, z4, z8, 0);
#else
  __builtin_amdgcn_tensor_load_to_lds(g0, g1, z4, z4, 0);
#endif
}

// ---------- problem constants ----------
#define NB 4
#define NSEQ 8192
#define DIMM 1024
#define NHEADS 16
#define DH 64
#define MROWS (NB * NSEQ)        // 32768
#define NQKV (3 * NHEADS * DH)   // 3072
#define NLEVELS 9

// ---------------------------------------------------------------
__global__ void convert_x_kernel(const float4* __restrict__ x,
                                 ushort4* __restrict__ xb, size_t n4) {
  size_t i = (size_t)blockIdx.x * blockDim.x + threadIdx.x;
  if (i >= n4) return;
  float4 v = x[i];
  xb[i] = make_ushort4(f2bf(v.x), f2bf(v.y), f2bf(v.z), f2bf(v.w));
}

__global__ void transpose_bf16_kernel(const float* __restrict__ w,
                                      unsigned short* __restrict__ wt,
                                      int K, int Ncols) {
  size_t idx = (size_t)blockIdx.x * blockDim.x + threadIdx.x;
  size_t total = (size_t)K * Ncols;
  if (idx >= total) return;
  int k = (int)(idx % K);
  size_t n = idx / K;
  wt[n * (size_t)K + k] = f2bf(w[(size_t)k * Ncols + n]);
}

// ---------------------------------------------------------------
// QKV GEMM with TDM-staged B tiles (double buffered in LDS).
// Block: 256 thr = 8 waves; wave tile 16x64; block tile 128x64; K-step 32.
__global__ void gemm_qkv_kernel(const unsigned short* __restrict__ xb,
                                const unsigned short* __restrict__ wt,
                                unsigned short* __restrict__ qh,
                                unsigned short* __restrict__ kh,
                                unsigned short* __restrict__ vh) {
  const int K = DIMM;
  __shared__ unsigned short ldsB[2][64 * 32];   // 2 x 4KB B tiles
  int wv = threadIdx.x >> 5, lane = threadIdx.x & 31;
  int ll = lane & 15, lh = lane >> 4;
  int m0 = blockIdx.x * 128 + wv * 16;
  int n0 = blockIdx.y * 64;

  const unsigned short* ap = xb + (size_t)(m0 + ll) * K + lh * 8;
  const unsigned short* bt = wt + (size_t)n0 * K;   // B tile rows start here
  unsigned lds0 = (unsigned)(uintptr_t)&ldsB[0][0];
  unsigned lds1 = (unsigned)(uintptr_t)&ldsB[1][0];

  // preload tile 0
  if (wv == 0) {
    tdm_load_2d(lds0, bt, K, NQKV, 32, 64, K);
    __builtin_amdgcn_s_wait_tensorcnt(0);
  }
  __syncthreads();

  v8f c[4] = {};
  for (int k0 = 0; k0 < K; k0 += 32) {
    int cur = (k0 >> 5) & 1;
    if (wv == 0 && (k0 + 32) < K)
      tdm_load_2d(cur ? lds0 : lds1, bt + (k0 + 32), K, NQKV, 32, 64, K);

    __builtin_prefetch(ap + k0 + 512, 0, 1);
    FragBF fa;
    fa.q[0] = *(const uint4*)(ap + k0);
    fa.q[1] = *(const uint4*)(ap + k0 + 16);
#pragma unroll
    for (int t = 0; t < 4; ++t) {
      FragBF fb;
      const unsigned short* bp = &ldsB[cur][(t * 16 + ll) * 32 + lh * 8];
      fb.q[0] = *(const uint4*)bp;
      fb.q[1] = *(const uint4*)(bp + 16);
      c[t] = wmma_bf16(fa, fb, c[t]);
    }
    if (wv == 0) __builtin_amdgcn_s_wait_tensorcnt(0);  // next tile landed
    __syncthreads();
  }

  // scatter: col -> (which, head, d); row -> (batch, n)
#pragma unroll
  for (int t = 0; t < 4; ++t) {
    int col = n0 + t * 16 + ll;
    int which = col >> 10, rem = col & 1023;
    int h = rem >> 6, d = rem & 63;
    unsigned short* dst = (which == 0) ? qh : ((which == 1) ? kh : vh);
    float scl = (which == 0) ? 0.125f : 1.0f;   // 64^-0.5
#pragma unroll
    for (int i = 0; i < 8; ++i) {
      int r = m0 + 8 * lh + i;
      int bi = r >> 13, n = r & (NSEQ - 1);
      dst[(((size_t)(bi * NHEADS + h)) * NSEQ + n) * DH + d] = f2bf(c[t][i] * scl);
    }
  }
}

// ---------------------------------------------------------------
__global__ void pool_kernel(const unsigned short* __restrict__ qi,
                            const unsigned short* __restrict__ ki,
                            const unsigned short* __restrict__ vi,
                            unsigned short* __restrict__ qo,
                            unsigned short* __restrict__ ko,
                            unsigned short* __restrict__ vo, int Nout) {
  size_t idx = (size_t)blockIdx.x * blockDim.x + threadIdx.x;
  size_t total = (size_t)64 * Nout * DH;
  if (idx >= total) return;
  int d = (int)(idx & 63);
  size_t t = idx >> 6;
  int m = (int)(t % Nout);
  int bh = (int)(t / Nout);
  size_t i0 = (((size_t)bh * (2 * Nout)) + 2 * m) * DH + d;
  size_t i1 = i0 + DH;
  qo[idx] = f2bf(0.5f * (bf2f(qi[i0]) + bf2f(qi[i1])));
  ko[idx] = f2bf(0.5f * (bf2f(ki[i0]) + bf2f(ki[i1])));
  vo[idx] = f2bf(bf2f(vi[i0]) + bf2f(vi[i1]));
}

// ---------------------------------------------------------------
// One wave per 16-row block: S = q k^T (2 WMMA, K=64), rowmax/exp/rowsum via
// 16-lane shuffles, P through LDS, y = P V (4 WMMA, K padded 16->32).
__global__ void attn_level_kernel(const unsigned short* __restrict__ q,
                                  const unsigned short* __restrict__ k,
                                  const unsigned short* __restrict__ v,
                                  float* __restrict__ y,
                                  float* __restrict__ a,
                                  int Nl, int flip) {
  __shared__ float pbuf[8][16][17];
  int wv = threadIdx.x >> 5, lane = threadIdx.x & 31;
  int ll = lane & 15, lh = lane >> 4;
  int nb = Nl >> 4;
  int task = blockIdx.x * 8 + wv;
  int bh = task / nb, blk = task % nb;
  int kblk = flip ? (blk ^ 1) : blk;

  const unsigned short* qb = q + (((size_t)bh * Nl) + blk * 16) * DH;
  const unsigned short* kb = k + (((size_t)bh * Nl) + kblk * 16) * DH;
  const unsigned short* vb = v + (((size_t)bh * Nl) + kblk * 16) * DH;

  v8f s = {};
#pragma unroll
  for (int kc = 0; kc < 2; ++kc) {
    FragBF fa, fb;
    const unsigned short* ap = qb + ll * DH + kc * 32 + lh * 8;
    fa.q[0] = *(const uint4*)ap;
    fa.q[1] = *(const uint4*)(ap + 16);
    const unsigned short* bp = kb + ll * DH + kc * 32 + lh * 8;
    fb.q[0] = *(const uint4*)bp;
    fb.q[1] = *(const uint4*)(bp + 16);
    s = wmma_bf16(fa, fb, s);
  }

  float p[8], rs[8];
#pragma unroll
  for (int i = 0; i < 8; ++i) {
    float mx = s[i];
#pragma unroll
    for (int msk = 1; msk < 16; msk <<= 1) mx = fmaxf(mx, __shfl_xor(mx, msk, 32));
    float e = __expf(s[i] - mx);
    p[i] = e;
    float r = e;
#pragma unroll
    for (int msk = 1; msk < 16; msk <<= 1) r += __shfl_xor(r, msk, 32);
    rs[i] = r;
  }

#pragma unroll
  for (int i = 0; i < 8; ++i) pbuf[wv][8 * lh + i][ll] = p[i];
  asm volatile("s_wait_dscnt 0" ::: "memory");

  FragBF fp;
#pragma unroll
  for (int e = 0; e < 8; ++e) fp.u[e] = f2bf(pbuf[wv][ll][lh * 8 + e]);
#pragma unroll
  for (int e = 8; e < 16; ++e) fp.u[e] = 0;

#pragma unroll
  for (int t = 0; t < 4; ++t) {
    FragBF fv;
#pragma unroll
    for (int e = 0; e < 16; ++e) fv.u[e] = 0;
    {
#pragma unroll
      for (int e = 0; e < 8; ++e) fv.u[e] = vb[(size_t)(lh * 8 + e) * DH + t * 16 + ll];
    }
    v8f yc = {};
    yc = wmma_bf16(fp, fv, yc);
#pragma unroll
    for (int i = 0; i < 8; ++i)
      y[(((size_t)bh * Nl) + blk * 16 + 8 * lh + i) * DH + t * 16 + ll] = yc[i];
  }
  if (ll == 0) {
#pragma unroll
    for (int i = 0; i < 8; ++i)
      a[((size_t)bh * Nl) + blk * 16 + 8 * lh + i] = rs[i];
  }
}

// ---------------------------------------------------------------
__global__ void combine_kernel(const float* __restrict__ yb,
                               const float* __restrict__ ab,
                               unsigned short* __restrict__ attnb) {
  size_t idx = (size_t)blockIdx.x * 256 + threadIdx.x;
  int d = (int)(idx & 63);
  size_t t = idx >> 6;
  int n = (int)(t & (NSEQ - 1));
  int bh = (int)(t >> 13);
  float Y = 0.f, A = 0.f;
  size_t yo = 0, ao = 0;
#pragma unroll
  for (int l = 0; l < NLEVELS; ++l) {
    int Nl = NSEQ >> l;
    Y += yb[yo + (((size_t)bh * Nl) + (n >> l)) * DH + d];
    A += ab[ao + (size_t)bh * Nl + (n >> l)];
    yo += (size_t)64 * Nl * DH;
    ao += (size_t)64 * Nl;
  }
  float o = Y / (A + 1e-8f);
  int bi = bh >> 4, h = bh & 15;
  attnb[((size_t)bi * NSEQ + n) * DIMM + h * DH + d] = f2bf(o);
}

// ---------------------------------------------------------------
// out = attnb(bf16) @ w_out + b_out  (f32), same TDM-staged B pipeline.
__global__ void gemm_out_kernel(const unsigned short* __restrict__ ab,
                                const unsigned short* __restrict__ wt,
                                const float* __restrict__ bias,
                                float* __restrict__ out) {
  const int K = DIMM;
  __shared__ unsigned short ldsB[2][64 * 32];
  int wv = threadIdx.x >> 5, lane = threadIdx.x & 31;
  int ll = lane & 15, lh = lane >> 4;
  int m0 = blockIdx.x * 128 + wv * 16;
  int n0 = blockIdx.y * 64;

  const unsigned short* ap = ab + (size_t)(m0 + ll) * K + lh * 8;
  const unsigned short* bt = wt + (size_t)n0 * K;
  unsigned lds0 = (unsigned)(uintptr_t)&ldsB[0][0];
  unsigned lds1 = (unsigned)(uintptr_t)&ldsB[1][0];

  if (wv == 0) {
    tdm_load_2d(lds0, bt, K, DIMM, 32, 64, K);
    __builtin_amdgcn_s_wait_tensorcnt(0);
  }
  __syncthreads();

  v8f c[4] = {};
  for (int k0 = 0; k0 < K; k0 += 32) {
    int cur = (k0 >> 5) & 1;
    if (wv == 0 && (k0 + 32) < K)
      tdm_load_2d(cur ? lds0 : lds1, bt + (k0 + 32), K, DIMM, 32, 64, K);

    __builtin_prefetch(ap + k0 + 512, 0, 1);
    FragBF fa;
    fa.q[0] = *(const uint4*)(ap + k0);
    fa.q[1] = *(const uint4*)(ap + k0 + 16);
#pragma unroll
    for (int t = 0; t < 4; ++t) {
      FragBF fb;
      const unsigned short* bp = &ldsB[cur][(t * 16 + ll) * 32 + lh * 8];
      fb.q[0] = *(const uint4*)bp;
      fb.q[1] = *(const uint4*)(bp + 16);
      c[t] = wmma_bf16(fa, fb, c[t]);
    }
    if (wv == 0) __builtin_amdgcn_s_wait_tensorcnt(0);
    __syncthreads();
  }

#pragma unroll
  for (int t = 0; t < 4; ++t) {
    int col = n0 + t * 16 + ll;
    float b = bias[col];
#pragma unroll
    for (int i = 0; i < 8; ++i) {
      int r = m0 + 8 * lh + i;
      out[(size_t)r * DIMM + col] = c[t][i] + b;
    }
  }
}

// ---------------------------------------------------------------
extern "C" void kernel_launch(void* const* d_in, const int* in_sizes, int n_in,
                              void* d_out, int out_size, void* d_ws, size_t ws_size,
                              hipStream_t stream) {
  (void)in_sizes; (void)n_in; (void)out_size; (void)ws_size;
  const float* x     = (const float*)d_in[0];
  const float* w_qkv = (const float*)d_in[1];
  const float* w_out = (const float*)d_in[2];
  const float* b_out = (const float*)d_in[3];
  float* out = (float*)d_out;

  int    lens[NLEVELS];
  size_t off[NLEVELS], aoff[NLEVELS];
  size_t o = 0, ao = 0;
  for (int l = 0; l < NLEVELS; ++l) {
    int Nl = NSEQ >> l;
    lens[l] = Nl; off[l] = o; aoff[l] = ao;
    o  += (size_t)64 * Nl * DH;
    ao += (size_t)64 * Nl;
  }
  const size_t LVL_TOT = o;
  const size_t A_TOT   = ao;

  char* p = (char*)d_ws;
  unsigned short* xb    = (unsigned short*)p; p += (size_t)MROWS * DIMM * 2;
  unsigned short* wqkvt = (unsigned short*)p; p += (size_t)NQKV * DIMM * 2;
  unsigned short* woutt = (unsigned short*)p; p += (size_t)DIMM * DIMM * 2;
  unsigned short* qh    = (unsigned short*)p; p += LVL_TOT * 2;
  unsigned short* kh    = (unsigned short*)p; p += LVL_TOT * 2;
  unsigned short* vh    = (unsigned short*)p; p += LVL_TOT * 2;
  float* yb             = (float*)p;          p += LVL_TOT * 4;
  float* ab             = (float*)p;          p += A_TOT * 4;
  unsigned short* attnb = (unsigned short*)p; p += (size_t)MROWS * DIMM * 2;

  {
    size_t n4 = (size_t)MROWS * DIMM / 4;
    convert_x_kernel<<<(unsigned)((n4 + 255) / 256), 256, 0, stream>>>(
        (const float4*)x, (ushort4*)xb, n4);
    size_t t1 = (size_t)DIMM * NQKV;
    transpose_bf16_kernel<<<(unsigned)((t1 + 255) / 256), 256, 0, stream>>>(
        w_qkv, wqkvt, DIMM, NQKV);
    size_t t2 = (size_t)DIMM * DIMM;
    transpose_bf16_kernel<<<(unsigned)((t2 + 255) / 256), 256, 0, stream>>>(
        w_out, woutt, DIMM, DIMM);
  }

  gemm_qkv_kernel<<<dim3(MROWS / 128, NQKV / 64), 256, 0, stream>>>(
      xb, wqkvt, qh, kh, vh);

  for (int l = 1; l < NLEVELS; ++l) {
    int Nout = lens[l];
    size_t total = (size_t)64 * Nout * DH;
    pool_kernel<<<(unsigned)((total + 255) / 256), 256, 0, stream>>>(
        qh + off[l - 1], kh + off[l - 1], vh + off[l - 1],
        qh + off[l],     kh + off[l],     vh + off[l], Nout);
  }

  for (int l = 0; l < NLEVELS; ++l) {
    int Nl = lens[l];
    int nb = Nl / 16;
    unsigned blocks = (unsigned)((64 * nb) / 8);
    attn_level_kernel<<<blocks, 256, 0, stream>>>(
        qh + off[l], kh + off[l], vh + off[l],
        yb + off[l], ab + aoff[l], Nl, (l != 0) ? 1 : 0);
  }

  {
    size_t total = (size_t)64 * NSEQ * DH;
    combine_kernel<<<(unsigned)(total / 256), 256, 0, stream>>>(yb, ab, attnb);
  }

  gemm_out_kernel<<<dim3(MROWS / 128, DIMM / 64), 256, 0, stream>>>(
      attnb, woutt, b_out, out);
}
